// MTL_Decoder_39865886442144
// MI455X (gfx1250) — matrix operand order repaired and checked by reading, not compiled, not run on hardware
//
#include <hip/hip_runtime.h>
#include <hip/hip_bf16.h>
#include <math.h>

typedef __bf16 bf16;
typedef __bf16 v16bf __attribute__((ext_vector_type(16)));
typedef float  v8f   __attribute__((ext_vector_type(8)));

union BFrag { v16bf v; bf16 h[16]; uint4 q[2]; };

#define V8F_ZERO {0.f, 0.f, 0.f, 0.f, 0.f, 0.f, 0.f, 0.f}

// ---- WMMA wrapper: D = A(16x32 bf16) x B(32x16 bf16) + C(16x16 f32) ----
__device__ __forceinline__ v8f wmma_bf16(v16bf a, v16bf b, v8f c) {
    return __builtin_amdgcn_wmma_f32_16x16x32_bf16(false, a, false, b, (short)0, c, false, false);
}

// A-fragment, row-major bf16 source. `row` points at element [m][k_block*32],
// m = lane&15. Per-lane elements: k = half*8 + 0..7 and 16 + half*8 + 0..7.
__device__ __forceinline__ void load_a(BFrag& f, const bf16* row, int lane) {
    int half = (lane >> 4) & 1;
    f.q[0] = *(const uint4*)(row + half * 8);
    f.q[1] = *(const uint4*)(row + 16 + half * 8);
}

// A-fragment with K padded: real k = 0..15 only (upper 16 of K are zero).
__device__ __forceinline__ void load_a_klo(BFrag& f, const bf16* row, int lane) {
    int half = (lane >> 4) & 1;
    f.q[0] = *(const uint4*)(row + half * 8);
    f.q[1] = make_uint4(0u, 0u, 0u, 0u);
}

// A-fragment converting from f32 row-major source.
__device__ __forceinline__ void load_a_f32(BFrag& f, const float* row, int lane) {
    int half = (lane >> 4) & 1;
    const float* p0 = row + half * 8;
    const float* p1 = row + 16 + half * 8;
#pragma unroll
    for (int j = 0; j < 8; ++j) { f.h[j] = (bf16)p0[j]; f.h[8 + j] = (bf16)p1[j]; }
}

// B-fragment, column-major bf16 source. `col` points at element [n][k_block*32]
// of a [Ncols][K] array, n = lane&15. Per-lane: k = half*8 + 0..7 and 16 + half*8 + 0..7.
__device__ __forceinline__ void load_b(BFrag& f, const bf16* col, int lane) {
    int half = (lane >> 4) & 1;
    f.q[0] = *(const uint4*)(col + half * 8);
    f.q[1] = *(const uint4*)(col + 16 + half * 8);
}

// B-fragment with K padded to 32: valid k = 0..15 (elements 8..15 of each lane are zeros).
__device__ __forceinline__ void load_b_klo(BFrag& f, const bf16* col, int lane) {
    int half = (lane >> 4) & 1;
    f.q[0] = *(const uint4*)(col + half * 8);
    f.q[1] = make_uint4(0u, 0u, 0u, 0u);
}

// ---------------- constants ----------------
#define BB   64
#define PP   100
#define NN   1000
#define NPAD 1008
#define NV   1024   // Vt padded so the attn*V K-loop is tile aligned
#define EMB  256
#define HH   16
#define DD   16
#define CTX  1024
#define KQ   1284
#define KQP  1312   // 41 * 32
#define PPAD 112    // 7 * 16

// ---------------- conversion kernels ----------------
__global__ void k_convert_nodes(const float* __restrict__ src, bf16* __restrict__ dst) {
    size_t total = (size_t)BB * NPAD * EMB;
    for (size_t i = (size_t)blockIdx.x * blockDim.x + threadIdx.x; i < total;
         i += (size_t)gridDim.x * blockDim.x) {
        int e = (int)(i % EMB);
        size_t t = i / EMB;
        int n = (int)(t % NPAD);
        int b = (int)(t / NPAD);
        float v = (n < NN) ? src[((size_t)b * NN + n) * EMB + e] : 0.f;
        dst[i] = (bf16)v;
    }
}

// dst[c][k] = (k < K) ? src[k][c] : 0 ;  dst is [C][Kpad] bf16
__global__ void k_transpose_w(const float* __restrict__ src, bf16* __restrict__ dst,
                              int K, int Kpad, int C) {
    size_t total = (size_t)C * Kpad;
    for (size_t i = (size_t)blockIdx.x * blockDim.x + threadIdx.x; i < total;
         i += (size_t)gridDim.x * blockDim.x) {
        int k = (int)(i % Kpad);
        int c = (int)(i / Kpad);
        dst[i] = (bf16)((k < K) ? src[(size_t)k * C + c] : 0.f);
    }
}

__global__ void k_zero_bf16(bf16* __restrict__ p, size_t n) {
    for (size_t i = (size_t)blockIdx.x * blockDim.x + threadIdx.x; i < n;
         i += (size_t)gridDim.x * blockDim.x)
        p[i] = (bf16)0.f;
}

// ---------------- K/V projection ----------------
// grid (63, B), 256 threads (8 waves). K -> kbf [B][NPAD][256] bf16,
// V -> vt [B][H][D][NV] bf16 (transposed, n-contiguous).
// Each wave processes the K-tile and V-tile of the same column block together,
// sharing the A fragment and double-buffering both B streams.
__global__ void __launch_bounds__(256) k_kv(const bf16* __restrict__ nodes,
                                            const bf16* __restrict__ wkt,
                                            const bf16* __restrict__ wvt,
                                            bf16* __restrict__ kbf,
                                            bf16* __restrict__ vt) {
    int nt = blockIdx.x, b = blockIdx.y;
    int n0 = nt * 16;
    int lane = threadIdx.x & 31, wid = threadIdx.x >> 5;
    int m = lane & 15, half = (lane >> 4) & 1;
    const bf16* arow = nodes + ((size_t)b * NPAD + n0 + m) * EMB;

#pragma unroll
    for (int pass = 0; pass < 2; ++pass) {
        int ct = wid + 8 * pass;           // 0..15
        int col = ct * 16 + m;
        const bf16* kc = wkt + (size_t)col * EMB;
        const bf16* vc = wvt + (size_t)col * EMB;
        BFrag kA, kB, vA, vB, a0, a1;
        load_b(kA, kc, lane);
        load_b(vA, vc, lane);
        v8f accK = V8F_ZERO, accV = V8F_ZERO;
#pragma unroll
        for (int kb = 0; kb < 8; kb += 2) {
            load_b(kB, kc + (kb + 1) * 32, lane);
            load_b(vB, vc + (kb + 1) * 32, lane);
            load_a(a0, arow + kb * 32, lane);
            accK = wmma_bf16(a0.v, kA.v, accK);
            accV = wmma_bf16(a0.v, vA.v, accV);
            if (kb + 2 < 8) {
                load_b(kA, kc + (kb + 2) * 32, lane);
                load_b(vA, vc + (kb + 2) * 32, lane);
            }
            load_a(a1, arow + (kb + 1) * 32, lane);
            accK = wmma_bf16(a1.v, kB.v, accK);
            accV = wmma_bf16(a1.v, vB.v, accV);
        }
        // K store: [B][NPAD][256]
#pragma unroll
        for (int r = 0; r < 8; ++r) {
            int mm = r + 8 * half;
            kbf[((size_t)b * NPAD + n0 + mm) * EMB + col] = (bf16)accK[r];
        }
        // V store transposed: [B][H][D][NV], contiguous in n
        {
            int h = ct, d = m;
            bf16* p = vt + ((((size_t)b * HH + h) * DD + d) * NV) + n0 + 8 * half;
#pragma unroll
            for (int r = 0; r < 8; ++r) p[r] = (bf16)accV[r];
        }
    }
}

// ---------------- Q projection ----------------
// grid (7, B), 256 threads. qbuf [B][PPAD][256] bf16, pre-scaled by 1/sqrt(D).
// Each wave handles two column tiles concurrently with shared A fragments.
__global__ void __launch_bounds__(256) k_q(const float* __restrict__ eln,
                                           const float* __restrict__ attr,
                                           const float* __restrict__ mlc,
                                           const bf16* __restrict__ wqt,
                                           bf16* __restrict__ qbuf) {
    __shared__ __align__(16) bf16 As[16][KQP];
    int pt = blockIdx.x, b = blockIdx.y, p0 = pt * 16;
    int tid = threadIdx.x;

    for (int i = tid; i < 16 * KQP; i += 256) {
        int pl = i / KQP, c = i % KQP;
        int p = p0 + pl;
        float v = 0.f;
        if (p < PP) {
            if (c < 256)       v = eln[((size_t)b * PP + p) * EMB + c];
            else if (c < 260)  v = attr[((size_t)b * PP + p) * 4 + (c - 256)];
            else if (c < KQ)   v = mlc[(size_t)b * CTX + (c - 260)];
        }
        As[pl][c] = (bf16)v;
    }
    __syncthreads();

    int lane = tid & 31, wid = tid >> 5;
    int m = lane & 15, half = (lane >> 4) & 1;
    int col0 = wid * 16 + m;
    int col1 = (wid + 8) * 16 + m;
    const bf16* c0 = wqt + (size_t)col0 * KQP;
    const bf16* c1 = wqt + (size_t)col1 * KQP;

    BFrag x0A, x0B, x1A, x1B, a0, a1;
    load_b(x0A, c0, lane);
    load_b(x1A, c1, lane);
    v8f acc0 = V8F_ZERO, acc1 = V8F_ZERO;
    // 41 K-blocks: step-2 loop covers 0..39 (prefetch indices kb+1<=39, kb+2<=40
    // are always valid), then one tail block.
#pragma unroll
    for (int kb = 0; kb < 40; kb += 2) {
        load_b(x0B, c0 + (kb + 1) * 32, lane);
        load_b(x1B, c1 + (kb + 1) * 32, lane);
        load_a(a0, &As[m][kb * 32], lane);
        acc0 = wmma_bf16(a0.v, x0A.v, acc0);
        acc1 = wmma_bf16(a0.v, x1A.v, acc1);
        load_b(x0A, c0 + (kb + 2) * 32, lane);
        load_b(x1A, c1 + (kb + 2) * 32, lane);
        load_a(a1, &As[m][(kb + 1) * 32], lane);
        acc0 = wmma_bf16(a1.v, x0B.v, acc0);
        acc1 = wmma_bf16(a1.v, x1B.v, acc1);
    }
    load_a(a0, &As[m][40 * 32], lane);
    acc0 = wmma_bf16(a0.v, x0A.v, acc0);
    acc1 = wmma_bf16(a0.v, x1A.v, acc1);

#pragma unroll
    for (int r = 0; r < 8; ++r) {
        int mm = r + 8 * half;
        qbuf[((size_t)b * PPAD + p0 + mm) * EMB + col0] = (bf16)(0.25f * acc0[r]);
        qbuf[((size_t)b * PPAD + p0 + mm) * EMB + col1] = (bf16)(0.25f * acc1[r]);
    }
}

// ---------------- masked multi-head attention ----------------
// grid (7, H, B), 128 threads (4 waves).
__global__ void __launch_bounds__(128) k_attn(const bf16* __restrict__ qbuf,
                                              const bf16* __restrict__ kbf,
                                              const bf16* __restrict__ vt,
                                              const float* __restrict__ mask,
                                              float* __restrict__ attout) {
    __shared__ __align__(16) bf16 sbuf[16][NV];   // scores then weights (bf16)
    __shared__ float red[16][8];
    __shared__ float red2[16][8];
    __shared__ float outred[4][16][16];
    int pt = blockIdx.x, h = blockIdx.y, b = blockIdx.z;
    int p0 = pt * 16;
    int tid = threadIdx.x, lane = tid & 31, wid = tid >> 5;
    int m = lane & 15, half = (lane >> 4) & 1;

    // pad columns 1008..1023 with -inf
    for (int i = tid; i < 16 * 16; i += 128)
        sbuf[i >> 4][NPAD + (i & 15)] = (bf16)(-INFINITY);

    // q A-fragment (K padded 16 -> 32), scale already folded in
    BFrag aq;
    load_a_klo(aq, qbuf + ((size_t)b * PPAD + p0 + m) * EMB + h * 16, lane);

    // score tiles: prefetch the next tile's K fragment before the current WMMA
    BFrag bkcur, bknxt;
    load_b_klo(bkcur, kbf + ((size_t)b * NPAD + wid * 16 + m) * EMB + h * 16, lane);
    for (int nt = wid; nt < 63; nt += 4) {
        int ntn = nt + 4;
        if (ntn < 63)
            load_b_klo(bknxt, kbf + ((size_t)b * NPAD + ntn * 16 + m) * EMB + h * 16, lane);
        v8f acc = V8F_ZERO;
        acc = wmma_bf16(aq.v, bkcur.v, acc);
        int n = nt * 16 + m;
#pragma unroll
        for (int r = 0; r < 8; ++r) {
            int prow = r + 8 * half;
            int pp = p0 + prow;
            float val;
            if (n >= NN) {
                val = -INFINITY;
            } else {
                val = acc[r];
                if (pp < PP) val += mask[((size_t)b * PP + pp) * NN + n];
            }
            sbuf[prow][n] = (bf16)val;
        }
        bkcur = bknxt;
    }
    __syncthreads();

    // softmax over n (8 threads per row), in-place in bf16 buffer
    {
        int r = tid >> 3, c = tid & 7;
        float mx = -INFINITY;
        for (int n = c; n < NV; n += 8) mx = fmaxf(mx, (float)sbuf[r][n]);
        red[r][c] = mx;
        __syncthreads();
        mx = red[r][0];
#pragma unroll
        for (int i = 1; i < 8; ++i) mx = fmaxf(mx, red[r][i]);
        float sum = 0.f;
        for (int n = c; n < NV; n += 8) {
            float e = __expf((float)sbuf[r][n] - mx);
            sum += e;
            sbuf[r][n] = (bf16)e;
        }
        red2[r][c] = sum;
        __syncthreads();
        sum = 0.f;
#pragma unroll
        for (int i = 0; i < 8; ++i) sum += red2[r][i];
        float inv = 1.f / sum;
        for (int n = c; n < NV; n += 8) sbuf[r][n] = (bf16)((float)sbuf[r][n] * inv);
    }
    __syncthreads();

    // out = w @ V : K = 1024 split over 4 waves (8 blocks each), reduce in LDS.
    // V fragments double-buffered.
    {
        const bf16* vbase = vt + (((size_t)b * HH + h) * DD + m) * NV;  // d = lane&15
        BFrag vA, vB, a0, a1;
        load_b(vA, vbase + (wid * 8) * 32, lane);
        v8f acc = V8F_ZERO;
#pragma unroll
        for (int kk = 0; kk < 8; kk += 2) {
            int kblk = wid * 8 + kk;
            load_b(vB, vbase + (kblk + 1) * 32, lane);
            load_a(a0, &sbuf[m][kblk * 32], lane);
            acc = wmma_bf16(a0.v, vA.v, acc);
            if (kk + 2 < 8) load_b(vA, vbase + (kblk + 2) * 32, lane);
            load_a(a1, &sbuf[m][(kblk + 1) * 32], lane);
            acc = wmma_bf16(a1.v, vB.v, acc);
        }
#pragma unroll
        for (int r = 0; r < 8; ++r) outred[wid][r + 8 * half][m] = acc[r];
    }
    __syncthreads();
    for (int i = tid; i < 256; i += 128) {
        int mm = i >> 4, d = i & 15;
        float s = outred[0][mm][d] + outred[1][mm][d] + outred[2][mm][d] + outred[3][mm][d];
        attout[((size_t)b * PPAD + p0 + mm) * EMB + h * 16 + d] = s;
    }
}

// ---------------- combine + pointer logits + softmax ----------------
// grid (7, B), 128 threads (4 waves).
__global__ void __launch_bounds__(128) k_final(const float* __restrict__ attout,
                                               const bf16* __restrict__ wct,
                                               const float* __restrict__ bcomb,
                                               const bf16* __restrict__ nodes,
                                               const float* __restrict__ mask,
                                               float* __restrict__ out) {
    __shared__ __align__(16) bf16 mhs[16][EMB];
    __shared__ __align__(16) bf16 sbuf[16][NV];
    __shared__ float red[16][8];
    __shared__ float red2[16][8];
    int pt = blockIdx.x, b = blockIdx.y, p0 = pt * 16;
    int tid = threadIdx.x, lane = tid & 31, wid = tid >> 5;
    int m = lane & 15, half = (lane >> 4) & 1;

    for (int i = tid; i < 256; i += 128)
        sbuf[i >> 4][NPAD + (i & 15)] = (bf16)(-INFINITY);

    const float* arow = attout + ((size_t)b * PPAD + p0 + m) * EMB;

    // stage 1: mh = attout @ Wcomb + bcomb -> LDS (bf16)
    // two passes, each with two column-tile streams sharing the A fragment
#pragma unroll
    for (int pass = 0; pass < 2; ++pass) {
        int col0 = (wid + 8 * pass) * 16 + m;
        int col1 = (wid + 4 + 8 * pass) * 16 + m;
        const bf16* c0 = wct + (size_t)col0 * EMB;
        const bf16* c1 = wct + (size_t)col1 * EMB;
        BFrag x0A, x0B, x1A, x1B, a0, a1;
        load_b(x0A, c0, lane);
        load_b(x1A, c1, lane);
        v8f acc0 = V8F_ZERO, acc1 = V8F_ZERO;
#pragma unroll
        for (int kb = 0; kb < 8; kb += 2) {
            load_b(x0B, c0 + (kb + 1) * 32, lane);
            load_b(x1B, c1 + (kb + 1) * 32, lane);
            load_a_f32(a0, arow + kb * 32, lane);
            acc0 = wmma_bf16(a0.v, x0A.v, acc0);
            acc1 = wmma_bf16(a0.v, x1A.v, acc1);
            if (kb + 2 < 8) {
                load_b(x0A, c0 + (kb + 2) * 32, lane);
                load_b(x1A, c1 + (kb + 2) * 32, lane);
            }
            load_a_f32(a1, arow + (kb + 1) * 32, lane);
            acc0 = wmma_bf16(a1.v, x0B.v, acc0);
            acc1 = wmma_bf16(a1.v, x1B.v, acc1);
        }
        float b0 = bcomb[col0];
        float b1 = bcomb[col1];
#pragma unroll
        for (int r = 0; r < 8; ++r) {
            mhs[r + 8 * half][col0] = (bf16)(acc0[r] + b0);
            mhs[r + 8 * half][col1] = (bf16)(acc1[r] + b1);
        }
    }
    __syncthreads();

    // stage 2: pointer logits s = clip*tanh((mh . node)/16) + mask
    for (int nt = wid; nt < 63; nt += 4) {
        int n0 = nt * 16;
        const bf16* ncol = nodes + ((size_t)b * NPAD + n0 + m) * EMB;
        BFrag nA, nB, a0, a1;
        load_b(nA, ncol, lane);
        v8f acc = V8F_ZERO;
#pragma unroll
        for (int kb = 0; kb < 8; kb += 2) {
            load_b(nB, ncol + (kb + 1) * 32, lane);
            load_a(a0, &mhs[m][kb * 32], lane);
            acc = wmma_bf16(a0.v, nA.v, acc);
            if (kb + 2 < 8) load_b(nA, ncol + (kb + 2) * 32, lane);
            load_a(a1, &mhs[m][(kb + 1) * 32], lane);
            acc = wmma_bf16(a1.v, nB.v, acc);
        }
        int n = n0 + m;
#pragma unroll
        for (int r = 0; r < 8; ++r) {
            int prow = r + 8 * half;
            int pp = p0 + prow;
            float val;
            if (n >= NN) {
                val = -INFINITY;
            } else {
                val = 10.f * tanhf(acc[r] * 0.0625f);
                if (pp < PP) val += mask[((size_t)b * PP + pp) * NN + n];
            }
            sbuf[prow][n] = (bf16)val;
        }
    }
    __syncthreads();

    // softmax + write probs
    {
        int r = tid >> 3, c = tid & 7;
        float mx = -INFINITY;
        for (int n = c; n < NV; n += 8) mx = fmaxf(mx, (float)sbuf[r][n]);
        red[r][c] = mx;
        __syncthreads();
        mx = red[r][0];
#pragma unroll
        for (int i = 1; i < 8; ++i) mx = fmaxf(mx, red[r][i]);
        float sum = 0.f;
        for (int n = c; n < NV; n += 8) {
            float e = __expf((float)sbuf[r][n] - mx);
            sum += e;
            sbuf[r][n] = (bf16)e;
        }
        red2[r][c] = sum;
        __syncthreads();
        sum = 0.f;
#pragma unroll
        for (int i = 0; i < 8; ++i) sum += red2[r][i];
        float inv = 1.f / sum;
        int pp = p0 + r;
        if (pp < PP) {
            for (int n = c; n < NN; n += 8)
                out[((size_t)b * PP + pp) * NN + n] = (float)sbuf[r][n] * inv;
        }
    }
}

// ---------------- launcher ----------------
extern "C" void kernel_launch(void* const* d_in, const int* in_sizes, int n_in,
                              void* d_out, int out_size, void* d_ws, size_t ws_size,
                              hipStream_t stream) {
    const float* eln   = (const float*)d_in[0];  // [B,P,EMB]
    const float* attr  = (const float*)d_in[1];  // [B,P,4]
    const float* mlc   = (const float*)d_in[2];  // [B,CTX]
    const float* mask  = (const float*)d_in[3];  // [B,P,N]
    const float* enc   = (const float*)d_in[4];  // [B,N,EMB]
    const float* Wq    = (const float*)d_in[5];  // [1284,256]
    const float* Wk    = (const float*)d_in[6];  // [256,256]
    const float* Wv    = (const float*)d_in[7];  // [256,256]
    const float* Wc    = (const float*)d_in[8];  // [256,256]
    const float* bcomb = (const float*)d_in[9];  // [256]
    float* out = (float*)d_out;

    char* ws = (char*)d_ws;
    size_t off = 0;
    auto alloc = [&](size_t bytes) {
        size_t r = off;
        off += (bytes + 255) & ~(size_t)255;
        return r;
    };
    bf16* nodes_bf = (bf16*)(ws + alloc((size_t)BB * NPAD * EMB * 2));
    bf16* wkt      = (bf16*)(ws + alloc((size_t)EMB * EMB * 2));
    bf16* wvt      = (bf16*)(ws + alloc((size_t)EMB * EMB * 2));
    bf16* wqt      = (bf16*)(ws + alloc((size_t)EMB * KQP * 2));
    bf16* wct      = (bf16*)(ws + alloc((size_t)EMB * EMB * 2));
    bf16* kbf      = (bf16*)(ws + alloc((size_t)BB * NPAD * EMB * 2));
    bf16* vt       = (bf16*)(ws + alloc((size_t)BB * HH * DD * NV * 2));
    bf16* qbuf     = (bf16*)(ws + alloc((size_t)BB * PPAD * EMB * 2));
    float* attout  = (float*)(ws + alloc((size_t)BB * PPAD * EMB * 4));
    (void)in_sizes; (void)n_in; (void)out_size; (void)ws_size;

    k_convert_nodes<<<2048, 256, 0, stream>>>(enc, nodes_bf);
    k_transpose_w<<<256, 256, 0, stream>>>(Wk, wkt, 256, 256, 256);
    k_transpose_w<<<256, 256, 0, stream>>>(Wv, wvt, 256, 256, 256);
    k_transpose_w<<<256, 256, 0, stream>>>(Wq, wqt, KQ, KQP, 256);
    k_transpose_w<<<256, 256, 0, stream>>>(Wc, wct, 256, 256, 256);
    k_zero_bf16<<<1024, 256, 0, stream>>>(vt, (size_t)BB * HH * DD * NV);

    k_kv<<<dim3(63, BB), 256, 0, stream>>>(nodes_bf, wkt, wvt, kbf, vt);
    k_q<<<dim3(7, BB), 256, 0, stream>>>(eln, attr, mlc, wqt, qbuf);
    k_attn<<<dim3(7, HH, BB), 128, 0, stream>>>(qbuf, kbf, vt, mask, attout);
    k_final<<<dim3(7, BB), 128, 0, stream>>>(attout, wct, bcomb, nodes_bf, mask, out);
}
